// Model_SAD_Bad_81423989998073
// MI455X (gfx1250) — matrix operand order repaired and checked by reading, not compiled, not run on hardware
//
#include <hip/hip_runtime.h>
#include <hip/hip_bf16.h>
#include <math.h>

// ---------------------------------------------------------------------------
// Problem constants (from reference): B=32, T=2048, D=1024
//   M = B*T = 65536 rows, K = N = 1024
//   out = [ x (M*N f32) | probs (M f32) | loss (1 f32) ]
// ---------------------------------------------------------------------------
#define BB       32
#define TT       2048
#define DD       1024
#define MROWS    (BB * TT)            // 65536
#define MN       ((size_t)MROWS * DD) // 67108864
#define KEEP     0.3f

typedef _Float16 v16h __attribute__((ext_vector_type(16)));
typedef _Float16 h8   __attribute__((ext_vector_type(8)));
typedef float    v8f  __attribute__((ext_vector_type(8)));

// LDS strides (in halfs) padded to keep 16B alignment and dodge bank conflicts
#define LDA 40   // A tile row stride: 64 rows x 32 k
#define LDB 40   // B tile (transposed) row stride: 128 n-rows x 32 k

// ws layout (floats): [0,65536) probs_acc  [65536,66560) center_acc  [66560,66562) loss_acc
#define WS_PROBS   0
#define WS_CENTER  65536
#define WS_LOSS    66560
#define WS_FLOATS  66562

// ---------------------------------------------------------------------------
// 0) zero the accumulators in workspace (must run every call: graph-safe)
// ---------------------------------------------------------------------------
__global__ void sad_ws_zero(float* ws) {
    int i = blockIdx.x * blockDim.x + threadIdx.x;
    if (i < WS_FLOATS) ws[i] = 0.0f;
}

// ---------------------------------------------------------------------------
// 1) Fused GEMM: x = dropout(relu(inputs @ W_fc + b_fc))
//    + per-column sums of label-0 bags (for center)
//    + per-row partial dots with W_cls (for probs)
//    Block tile: 64(M) x 128(N), K-step 32, 8 waves (wave32) per block.
//    Each wave: 16(M) x 64(N) via 4x v_wmma_f32_16x16x32_f16 accumulators.
// ---------------------------------------------------------------------------
__global__ __launch_bounds__(256)
void sad_fused_gemm(const float* __restrict__ A,      // inputs [M,K]
                    const int*   __restrict__ labels, // [B]
                    const float* __restrict__ U,      // dropout noise [M,N]
                    const float* __restrict__ Wfc,    // [K,N]
                    const float* __restrict__ bfc,    // [N]
                    const float* __restrict__ Wcls,   // [N]
                    float*       __restrict__ xout,   // [M,N] -> d_out
                    float*       __restrict__ probs_acc,  // [M]
                    float*       __restrict__ center_acc) // [N]
{
    __shared__ __align__(16) _Float16 Ash[64 * LDA];
    __shared__ __align__(16) _Float16 Bsh[128 * LDB];
    __shared__ float colsum[128];

    const int tid    = threadIdx.x;
    const int lane   = tid & 31;
    const int wave   = tid >> 5;
    const int waveM  = wave & 3;     // 4 M-subtiles of 16
    const int waveN  = wave >> 2;    // 2 N-subtiles of 64
    const int half   = lane >> 4;    // lane group (K split for A, K split for B)
    const int l15    = lane & 15;

    const int blockN = blockIdx.x;   // 0..7   (N/128)
    const int blockM = blockIdx.y;   // 0..1023 (M/64)

    // one bag per block (T=2048 is a multiple of 64)
    const int bag   = (blockM * 64) / TT;
    const int label = labels[bag];

    if (tid < 128) colsum[tid] = 0.0f;

    v8f acc[4] = {};

    const int am = waveM * 16 + l15; // A fragment row within block tile

    for (int kt = 0; kt < DD / 32; ++kt) {
        // ---- stage A tile: 64 x 32 f32 -> f16 LDS (512 float4, 2/thread) ----
        #pragma unroll
        for (int j = 0; j < 2; ++j) {
            int fid = tid + j * 256;
            int row = fid >> 3;        // 0..63
            int c4  = fid & 7;         // 0..7 (float4 within 32 cols)
            const float4 av = *(const float4*)
                (&A[(size_t)(blockM * 64 + row) * DD + kt * 32 + c4 * 4]);
            _Float16* dst = &Ash[row * LDA + c4 * 4];
            dst[0] = (_Float16)av.x; dst[1] = (_Float16)av.y;
            dst[2] = (_Float16)av.z; dst[3] = (_Float16)av.w;
            if (kt + 1 < DD / 32)
                __builtin_prefetch(&A[(size_t)(blockM * 64 + row) * DD +
                                      (kt + 1) * 32 + c4 * 4], 0, 1);
        }
        // ---- stage B tile: 32 x 128 f32 -> f16 LDS transposed (4/thread) ----
        #pragma unroll
        for (int j = 0; j < 4; ++j) {
            int fid = tid + j * 256;
            int kr  = fid >> 5;        // 0..31
            int c4  = fid & 31;        // 0..31 (float4 within 128 cols)
            const float4 bv = *(const float4*)
                (&Wfc[(size_t)(kt * 32 + kr) * DD + blockN * 128 + c4 * 4]);
            Bsh[(c4 * 4 + 0) * LDB + kr] = (_Float16)bv.x;
            Bsh[(c4 * 4 + 1) * LDB + kr] = (_Float16)bv.y;
            Bsh[(c4 * 4 + 2) * LDB + kr] = (_Float16)bv.z;
            Bsh[(c4 * 4 + 3) * LDB + kr] = (_Float16)bv.w;
            if (kt + 1 < DD / 32)
                __builtin_prefetch(&Wfc[(size_t)((kt + 1) * 32 + kr) * DD +
                                        blockN * 128 + c4 * 4], 0, 1);
        }
        __syncthreads();

        // ---- A fragment: lane = M row; element i -> k = (i>>3)*16 + half*8 + (i&7)
        h8 a0 = *(const h8*)(&Ash[am * LDA + half * 8]);
        h8 a1 = *(const h8*)(&Ash[am * LDA + 16 + half * 8]);
        v16h af = __builtin_shufflevector(a0, a1,
            0,1,2,3,4,5,6,7,8,9,10,11,12,13,14,15);

        // ---- 4 WMMAs: B fragment lane = N col; element i -> k = (i>>3)*16 + half*8 + (i&7)
        #pragma unroll
        for (int nt = 0; nt < 4; ++nt) {
            int nl = waveN * 64 + nt * 16 + l15;
            h8 b0 = *(const h8*)(&Bsh[nl * LDB + half * 8]);
            h8 b1 = *(const h8*)(&Bsh[nl * LDB + 16 + half * 8]);
            v16h bf = __builtin_shufflevector(b0, b1,
                0,1,2,3,4,5,6,7,8,9,10,11,12,13,14,15);
            acc[nt] = __builtin_amdgcn_wmma_f32_16x16x32_f16(
                false, af, false, bf, (short)0, acc[nt], false, false);
        }
        __syncthreads();
    }

    // ------------------- epilogue -------------------
    float prow[8] = {0,0,0,0,0,0,0,0};

    #pragma unroll
    for (int nt = 0; nt < 4; ++nt) {
        int n_g  = blockN * 128 + waveN * 64 + nt * 16 + l15;
        float bi = bfc[n_g];
        float wc = Wcls[n_g];
        float cs = 0.0f;
        #pragma unroll
        for (int r = 0; r < 8; ++r) {
            int   m_g = blockM * 64 + waveM * 16 + half * 8 + r;
            float v   = acc[nt][r] + bi;
            v = fmaxf(v, 0.0f);
            float uu = U[(size_t)m_g * DD + n_g];
            v = (uu < KEEP) ? v * (1.0f / KEEP) : 0.0f;
            xout[(size_t)m_g * DD + n_g] = v;
            prow[r] += v * wc;
            cs      += v;
        }
        if (label == 0)
            atomicAdd(&colsum[waveN * 64 + nt * 16 + l15], cs);
    }

    // probs partial: reduce over the 16-lane N-group (lanes with identical row)
    #pragma unroll
    for (int r = 0; r < 8; ++r) {
        float p = prow[r];
        p += __shfl_xor(p, 8);
        p += __shfl_xor(p, 4);
        p += __shfl_xor(p, 2);
        p += __shfl_xor(p, 1);
        if (l15 == 0) {
            int m_g = blockM * 64 + waveM * 16 + half * 8 + r;
            atomicAdd(&probs_acc[m_g], p);
        }
    }

    __syncthreads();
    if (tid < 128 && label == 0)
        atomicAdd(&center_acc[blockN * 128 + tid], colsum[tid]);
}

// ---------------------------------------------------------------------------
// 2) probs = sigmoid(row_dot + b_cls)
// ---------------------------------------------------------------------------
__global__ void sad_probs_finalize(const float* __restrict__ probs_acc,
                                   const float* __restrict__ bcls,
                                   float* __restrict__ probs_out) {
    int i = blockIdx.x * blockDim.x + threadIdx.x;
    if (i < MROWS) {
        float v = probs_acc[i] + bcls[0];
        probs_out[i] = 1.0f / (1.0f + expf(-v));
    }
}

// ---------------------------------------------------------------------------
// 3) accumulate center_loss / sep_loss sums over T*D elements of x[last0]/x[last1]
// ---------------------------------------------------------------------------
__global__ __launch_bounds__(256)
void sad_loss_accum(const float* __restrict__ xout,
                    const int*   __restrict__ labels,
                    const float* __restrict__ center_acc,
                    float*       __restrict__ loss_acc) {
    // labels scan (trivial: 32 iterations)
    int last0 = -1, last1 = -1, n0 = 0;
    for (int b = 0; b < BB; ++b) {
        if (labels[b] == 0) { last0 = b; ++n0; }
        else                { last1 = b; }
    }
    const float inv_norm = 1.0f / ((float)n0 * (float)TT);

    int idx = blockIdx.x * blockDim.x + threadIdx.x;  // over T*D = 2097152
    int t = idx >> 10;
    int d = idx & (DD - 1);

    float c  = center_acc[d] * inv_norm;
    float x0 = xout[((size_t)(last0 * TT + t)) * DD + d];
    float x1 = xout[((size_t)(last1 * TT + t)) * DD + d];
    float d0 = x0 - c;
    float e0 = d0 * d0;
    float w  = x1 - c + 0.01f;
    float e1 = 1.0f / (w * w);

    __shared__ float s0[256], s1[256];
    int tid = threadIdx.x;
    s0[tid] = e0; s1[tid] = e1;
    __syncthreads();
    for (int s = 128; s > 0; s >>= 1) {
        if (tid < s) { s0[tid] += s0[tid + s]; s1[tid] += s1[tid + s]; }
        __syncthreads();
    }
    if (tid == 0) {
        atomicAdd(&loss_acc[0], s0[0]);
        atomicAdd(&loss_acc[1], s1[0]);
    }
}

// ---------------------------------------------------------------------------
// 4) combine into the scalar loss
// ---------------------------------------------------------------------------
__global__ void sad_loss_finalize(const float* __restrict__ loss_acc,
                                  float* __restrict__ loss_out) {
    float center_loss = (loss_acc[0] / (float)TT) * 0.5f / (1.0f + (float)BB);
    float sep_loss    = (loss_acc[1] / (float)TT) * 1e-7f / (1.0f + (float)BB);
    loss_out[0] = center_loss + sep_loss;
}

// ---------------------------------------------------------------------------
extern "C" void kernel_launch(void* const* d_in, const int* in_sizes, int n_in,
                              void* d_out, int out_size, void* d_ws, size_t ws_size,
                              hipStream_t stream) {
    const float* inputs = (const float*)d_in[0];
    const int*   labels = (const int*)  d_in[1];
    const float* u      = (const float*)d_in[2];
    const float* W_fc   = (const float*)d_in[3];
    const float* b_fc   = (const float*)d_in[4];
    const float* W_cls  = (const float*)d_in[5];
    const float* b_cls  = (const float*)d_in[6];

    float* out = (float*)d_out;
    float* ws  = (float*)d_ws;

    float* probs_acc  = ws + WS_PROBS;
    float* center_acc = ws + WS_CENTER;
    float* loss_acc   = ws + WS_LOSS;

    // 0) zero accumulators
    sad_ws_zero<<<(WS_FLOATS + 255) / 256, 256, 0, stream>>>(ws);

    // 1) fused GEMM + epilogue
    dim3 grid(DD / 128, MROWS / 64);   // 8 x 1024
    sad_fused_gemm<<<grid, 256, 0, stream>>>(
        inputs, labels, u, W_fc, b_fc, W_cls,
        out /* x at offset 0 */, probs_acc, center_acc);

    // 2) probs finalize
    sad_probs_finalize<<<MROWS / 256, 256, 0, stream>>>(
        probs_acc, b_cls, out + MN);

    // 3) loss accumulation over T*D elements
    sad_loss_accum<<<(TT * DD) / 256, 256, 0, stream>>>(
        out, labels, center_acc, loss_acc);

    // 4) scalar combine
    sad_loss_finalize<<<1, 1, 0, stream>>>(loss_acc, out + MN + MROWS);
}
